// FeedForwardNetwork_18588618457698
// MI455X (gfx1250) — matrix operand order repaired and checked by reading, not compiled, not run on hardware
//
#include <hip/hip_runtime.h>
#include <hip/hip_bf16.h>
#include <math.h>

// ---------------------------------------------------------------------------
// MI455X (gfx1250) implementation.
//   64 GFLOP of contractions -> v_wmma_f32_16x16x32_f16 (f16 operands, f32 acc)
//   ~0.7 GB HBM traffic -> ~30us memory floor at 23.3 TB/s; grid-stage g tensor
//   (763 MB f32) is kept entirely in LDS by fusing to_grid/silu/from_grid.
// ---------------------------------------------------------------------------

typedef _Float16 half_t;
typedef _Float16 v16h __attribute__((ext_vector_type(16)));
typedef _Float16 h8   __attribute__((ext_vector_type(8)));
typedef float    v8f  __attribute__((ext_vector_type(8)));

#define LMAX    6
#define NSH     49            // (LMAX+1)^2
#define CC      128
#define HH      256
#define NN      4096
#define NBA     182           // 14*13 grid points
#define NBA_PAD 192
#define NSH_PAD 64
#define EPSV    1e-5f

// ---- WMMA fragment loaders (wave32, 16x16x32 f16) -------------------------
__device__ __forceinline__ v16h cat8(h8 a, h8 b) {
  return __builtin_shufflevector(a, b, 0,1,2,3,4,5,6,7,8,9,10,11,12,13,14,15);
}
// A matrix (16xK tile): lane holds row = lane&15; halves j: K = (j/8)*16 + (lane/16)*8 + j%8
__device__ __forceinline__ v16h load_frag_a(const half_t* base, int stride) {
  int lane = threadIdx.x & 31;
  int r = lane & 15, hi = lane >> 4;
  const h8* p0 = reinterpret_cast<const h8*>(base + r * stride + hi * 8);
  const h8* p1 = reinterpret_cast<const h8*>(base + r * stride + 16 + hi * 8);
  return cat8(*p0, *p1);
}
// B matrix (Kx16 tile) stored column-major as (n, K) rows: lane holds col = lane&15;
// halves j: K = (j/8)*16 + (lane/16)*8 + j%8
__device__ __forceinline__ v16h load_frag_b(const half_t* base, int stride) {
  int lane = threadIdx.x & 31;
  int n = lane & 15, hi = lane >> 4;
  const h8* p0 = reinterpret_cast<const h8*>(base + n * stride + hi * 8);
  return cat8(p0[0], p0[2]);
}
__device__ __forceinline__ float silu_f(float x) { return x / (1.f + __expf(-x)); }
__device__ __forceinline__ int lidx(int m) { int l = 0; while ((l + 1) * (l + 1) <= m) ++l; return l; }

// ---- Kernel 0: convert weights to f16 -------------------------------------
__global__ void convert_weights(const float* __restrict__ w1, const float* __restrict__ gw,
                                const float* __restrict__ w2,
                                half_t* __restrict__ w1h, half_t* __restrict__ gwh,
                                half_t* __restrict__ w2h) {
  int i = blockIdx.x * blockDim.x + threadIdx.x;
  int stridetot = gridDim.x * blockDim.x;
  const int n1 = (LMAX + 1) * HH * CC, n2 = HH * CC, n3 = (LMAX + 1) * CC * HH;
  for (int t = i; t < n1; t += stridetot) w1h[t] = (half_t)w1[t];
  for (int t = i; t < n2; t += stridetot) gwh[t] = (half_t)gw[t];
  for (int t = i; t < n3; t += stridetot) w2h[t] = (half_t)w2[t];
}

// ---- Kernel 1: norms -> xn (f16), one block per sample --------------------
__global__ void __launch_bounds__(256) norm_kernel(const float* __restrict__ x,
    const float* __restrict__ ln0w, const float* __restrict__ ln0b,
    const float* __restrict__ affw, half_t* __restrict__ xnh) {
  int n = blockIdx.x;
  int t = threadIdx.x;
  const float* xb = x + (size_t)n * NSH * CC;
  __shared__ float red[256];
  __shared__ float red2[256];
  __shared__ float s_mu, s_rs, s_inv;

  // layernorm stats over l=0 row (128 ch)
  float v0 = (t < CC) ? xb[t] : 0.f;
  red[t] = v0; red2[t] = v0 * v0;
  __syncthreads();
  for (int s = 128; s > 0; s >>= 1) {
    if (t < s) { red[t] += red[t + s]; red2[t] += red2[t + s]; }
    __syncthreads();
  }
  if (t == 0) {
    float mu = red[0] / CC;
    float var = red2[0] / CC - mu * mu;
    s_mu = mu; s_rs = rsqrtf(var + EPSV);
  }
  __syncthreads();

  // balance-weighted power norm over rows 1..48
  float acc = 0.f;
  for (int idx = t; idx < (NSH - 1) * CC; idx += 256) {
    int i = idx / CC + 1, c = idx % CC;
    int l = lidx(i);
    float bal = 1.f / (float)((2 * l + 1) * LMAX);
    float xv = xb[i * CC + c];
    acc += bal * xv * xv;
  }
  red[t] = acc;
  __syncthreads();
  for (int s = 128; s > 0; s >>= 1) {
    if (t < s) red[t] += red[t + s];
    __syncthreads();
  }
  if (t == 0) s_inv = rsqrtf(red[0] / CC + EPSV);
  __syncthreads();

  if (t < CC) {
    float l0n = (v0 - s_mu) * s_rs * ln0w[t] + ln0b[t];
    xnh[(size_t)n * NSH * CC + t] = (half_t)l0n;
  }
  for (int idx = t; idx < (NSH - 1) * CC; idx += 256) {
    int i = idx / CC + 1, c = idx % CC;
    int l = lidx(i);
    float val = xb[i * CC + c] * s_inv * affw[(l - 1) * CC + c];
    xnh[((size_t)n * NSH + i) * CC + c] = (half_t)val;
  }
}

// ---- Kernel 2: h[n,m,:] = xn[n,m,:] @ W^T  (m=49 => gate) -----------------
__global__ void __launch_bounds__(256) h_kernel(const half_t* __restrict__ xnh,
    const half_t* __restrict__ w1h, const half_t* __restrict__ gwh,
    const float* __restrict__ b1, const float* __restrict__ gb,
    half_t* __restrict__ hh) {
  int n0 = blockIdx.x * 16;
  int m = blockIdx.y;                      // 0..49
  int msrc = (m < NSH) ? m : 0;            // gate reads xn row 0
  __shared__ half_t a_lds[16 * CC];
  for (int idx = threadIdx.x; idx < 16 * CC; idx += 256) {
    int r = idx / CC, c = idx % CC;
    a_lds[idx] = xnh[((size_t)(n0 + r) * NSH + msrc) * CC + c];
  }
  __syncthreads();

  const half_t* W = (m < NSH) ? (w1h + (size_t)lidx(m) * HH * CC) : gwh;
  int wave = threadIdx.x >> 5;
  int lane = threadIdx.x & 31, cl = lane & 15, hi = lane >> 4;

  v16h afr[4];
  for (int kc = 0; kc < 4; ++kc) afr[kc] = load_frag_a(a_lds + kc * 32, CC);

  for (int cti = 0; cti < 2; ++cti) {
    int o0 = (wave * 2 + cti) * 16;
    v8f acc = {};
    for (int kc = 0; kc < 4; ++kc) {
      v16h b = load_frag_b(W + (size_t)o0 * CC + kc * 32, CC);
      acc = __builtin_amdgcn_wmma_f32_16x16x32_f16(false, afr[kc], false, b,
                                                   (short)0, acc, false, false);
    }
    int o = o0 + cl;
    float bias = (m == 0) ? b1[o] : ((m == NSH) ? gb[o] : 0.f);
    for (int v = 0; v < 8; ++v) {
      int nr = n0 + v + 8 * hi;
      hh[((size_t)nr * (NSH + 1) + m) * HH + o] = (half_t)(acc[v] + bias);
    }
  }
}

// ---- Kernel 3: fused  ht = from_grid^T @ silu(to_grid @ h), one block per n
__global__ void __launch_bounds__(256) grid_kernel(const half_t* __restrict__ hh,
    const float* __restrict__ to_grid, const float* __restrict__ from_grid,
    half_t* __restrict__ hth) {
  extern __shared__ half_t sm[];
  half_t* tg  = sm;                 // (192 x 64)  to_grid, zero padded
  half_t* fgT = sm + 12288;         // (64 x 192)  from_grid transposed
  half_t* hT  = sm + 24576;         // (256 x 64)  h[n] transposed
  half_t* Gt  = sm + 40960;         // 8 waves x (16 x 192) silu(G) transposed
  int n = blockIdx.x;
  int t = threadIdx.x;

  for (int idx = t; idx < 40960; idx += 256) sm[idx] = (half_t)0.f;
  __syncthreads();
  for (int idx = t; idx < NBA * NSH; idx += 256) {
    int ba = idx / NSH, i = idx % NSH;
    tg[ba * NSH_PAD + i] = (half_t)to_grid[idx];
    fgT[i * NBA_PAD + ba] = (half_t)from_grid[idx];
  }
  for (int idx = t; idx < NSH * HH; idx += 256) {
    int i = idx / HH, c = idx % HH;
    hT[c * NSH_PAD + i] = hh[((size_t)n * (NSH + 1) + i) * HH + c];
  }
  __syncthreads();

  int wave = t >> 5, lane = t & 31, cl = lane & 15, hi = lane >> 4;
  half_t* gtw = Gt + wave * 16 * NBA_PAD;

  for (int cti = 0; cti < 2; ++cti) {
    int c0 = (wave * 2 + cti) * 16;
    // G(192x16) = to_grid(192x64) @ hT-cols, silu, store transposed (c,ba)
    for (int mt = 0; mt < 12; ++mt) {
      v8f acc = {};
      for (int kc = 0; kc < 2; ++kc) {
        v16h a = load_frag_a(tg + mt * 16 * NSH_PAD + kc * 32, NSH_PAD);
        v16h b = load_frag_b(hT + (size_t)c0 * NSH_PAD + kc * 32, NSH_PAD);
        acc = __builtin_amdgcn_wmma_f32_16x16x32_f16(false, a, false, b,
                                                     (short)0, acc, false, false);
      }
      h8 pk;
      for (int v = 0; v < 8; ++v) pk[v] = (half_t)silu_f(acc[v]);
      *reinterpret_cast<h8*>(gtw + cl * NBA_PAD + mt * 16 + hi * 8) = pk;
    }
    // ht(64x16) = from_grid^T(64x192) @ G(192x16)
    for (int mt = 0; mt < 4; ++mt) {
      v8f acc = {};
      for (int kc = 0; kc < 6; ++kc) {
        v16h a = load_frag_a(fgT + mt * 16 * NBA_PAD + kc * 32, NBA_PAD);
        v16h b = load_frag_b(gtw + kc * 32, NBA_PAD);
        acc = __builtin_amdgcn_wmma_f32_16x16x32_f16(false, a, false, b,
                                                     (short)0, acc, false, false);
      }
      for (int v = 0; v < 8; ++v) {
        int i = mt * 16 + v + 8 * hi;
        if (i < NSH) {
          int c = c0 + cl;
          float val = acc[v];
          if (i == 0)   // row 0 of h2 is silu(gate) (gate stored as row 49 of hh)
            val = silu_f((float)hh[((size_t)n * (NSH + 1) + NSH) * HH + c]);
          hth[((size_t)n * NSH + i) * HH + c] = (half_t)val;
        }
      }
    }
  }
}

// ---- Kernel 4: out[n,m,:] = h2[n,m,:] @ w2[l(m)]^T (+b2 for m=0) ----------
__global__ void __launch_bounds__(256) out_kernel(const half_t* __restrict__ hth,
    const half_t* __restrict__ w2h, const float* __restrict__ b2,
    float* __restrict__ out) {
  int n0 = blockIdx.x * 16;
  int m = blockIdx.y;                      // 0..48
  __shared__ half_t a_lds[16 * HH];
  for (int idx = threadIdx.x; idx < 16 * HH; idx += 256) {
    int r = idx / HH, k = idx % HH;
    a_lds[idx] = hth[((size_t)(n0 + r) * NSH + m) * HH + k];
  }
  __syncthreads();

  const half_t* W = w2h + (size_t)lidx(m) * CC * HH;
  int wave = threadIdx.x >> 5;
  int lane = threadIdx.x & 31, cl = lane & 15, hi = lane >> 4;
  int o0 = wave * 16;
  v8f acc = {};
  for (int kc = 0; kc < 8; ++kc) {
    v16h a = load_frag_a(a_lds + kc * 32, HH);
    v16h b = load_frag_b(W + (size_t)o0 * HH + kc * 32, HH);
    acc = __builtin_amdgcn_wmma_f32_16x16x32_f16(false, a, false, b,
                                                 (short)0, acc, false, false);
  }
  int o = o0 + cl;
  float bias = (m == 0) ? b2[o] : 0.f;
  for (int v = 0; v < 8; ++v) {
    int nr = n0 + v + 8 * hi;
    out[((size_t)nr * NSH + m) * CC + o] = acc[v] + bias;
  }
}

// ---------------------------------------------------------------------------
extern "C" void kernel_launch(void* const* d_in, const int* in_sizes, int n_in,
                              void* d_out, int out_size, void* d_ws, size_t ws_size,
                              hipStream_t stream) {
  (void)in_sizes; (void)n_in; (void)out_size; (void)ws_size;
  const float* x        = (const float*)d_in[0];
  const float* ln0_w    = (const float*)d_in[1];
  const float* ln0_b    = (const float*)d_in[2];
  const float* aff_w    = (const float*)d_in[3];
  const float* w1       = (const float*)d_in[4];
  const float* b1       = (const float*)d_in[5];
  const float* gate_w   = (const float*)d_in[6];
  const float* gate_b   = (const float*)d_in[7];
  const float* w2       = (const float*)d_in[8];
  const float* b2       = (const float*)d_in[9];
  const float* to_grid  = (const float*)d_in[10];
  const float* from_grid= (const float*)d_in[11];
  float* out = (float*)d_out;

  char* ws = (char*)d_ws;
  size_t off = 0;
  half_t* xnh = (half_t*)(ws + off); off += (size_t)NN * NSH * CC * 2;        // 51.4 MB
  half_t* hh  = (half_t*)(ws + off); off += (size_t)NN * (NSH + 1) * HH * 2;  // 104.9 MB
  half_t* hth = (half_t*)(ws + off); off += (size_t)NN * NSH * HH * 2;        // 102.8 MB
  half_t* w1h = (half_t*)(ws + off); off += (size_t)(LMAX + 1) * HH * CC * 2;
  half_t* gwh = (half_t*)(ws + off); off += (size_t)HH * CC * 2;
  half_t* w2h = (half_t*)(ws + off); off += (size_t)(LMAX + 1) * CC * HH * 2;

  convert_weights<<<512, 256, 0, stream>>>(w1, gate_w, w2, w1h, gwh, w2h);
  norm_kernel<<<NN, 256, 0, stream>>>(x, ln0_w, ln0_b, aff_w, xnh);
  h_kernel<<<dim3(NN / 16, NSH + 1), 256, 0, stream>>>(xnh, w1h, gwh, b1, gate_b, hh);
  grid_kernel<<<NN, 256, 131072, stream>>>(hh, to_grid, from_grid, hth);
  out_kernel<<<dim3(NN / 16, NSH), 256, 0, stream>>>(hth, w2h, b2, out);
}
